// MILAttentionEntityRepresentation_55198919688617
// MI455X (gfx1250) — compile-verified, matrix operand order checked by reading, not run on hardware
//
#include <hip/hip_runtime.h>
#include <hip/hip_bf16.h>

// ---------------------------------------------------------------------------
// Problem constants (from reference): B=8, M=256, E=128, K=16, H=1024,
// A=256, R=1024, RMS=512.  BE = B*E = 1024.  HP = H+RMS = 1536.
// ---------------------------------------------------------------------------
#define Bc   8
#define Mc   256
#define Ec   128
#define Kc   16
#define Hc   1024
#define Ac   256
#define Rc   1024
#define RMSc 512
#define BEc  1024
#define HPc  1536
#define NEGV (-1e25f)

typedef __bf16 bf16_t;
typedef __attribute__((ext_vector_type(16))) __bf16 v16bf;
typedef __attribute__((ext_vector_type(8)))  float  v8f;

// ---------------------------------------------------------------------------
// WMMA helpers (v_wmma_f32_16x16x32_bf16; wave32; D = A*B + C, f32 accum)
// A-matrix per ISA: lane 0-15 -> M=lane, lanes 16-31 -> M=lane-16 with K+8;
// VGPR v holds K-pair at k = (v&3)*2 + (lane>=16 ? 8:0) + (v>=4 ? 16:0).
// B-matrix is the dual: lane&15 selects N, same K packing.
// C/D: VGPR v, lane -> m = v + (lane>=16 ? 8:0), n = lane&15.
// All fragment loads below are 16 bf16 per lane in two runs of 8 -> 2x b128 loads.
// ---------------------------------------------------------------------------
__device__ __forceinline__ int frag_kidx(int lane, int v) {
    return ((v & 3) << 1) + (((lane >> 4) & 1) << 3) + ((v >> 2) << 4);
}

// element(sel, k) located at p[sel*ld + k]  (sel = M for A, N for B)
__device__ __forceinline__ v16bf frag_selmajor(const bf16_t* p, int ld, int kbase, int lane) {
    v16bf r;
    const bf16_t* q = p + (size_t)(lane & 15) * ld + kbase;
#pragma unroll
    for (int v = 0; v < 8; ++v) {
        int kk = frag_kidx(lane, v);
        r[2 * v]     = q[kk];
        r[2 * v + 1] = q[kk + 1];
    }
    return r;
}

__device__ __forceinline__ v8f wmma_bf16(v16bf a, v16bf b, v8f c) {
    return __builtin_amdgcn_wmma_f32_16x16x32_bf16(false, a, false, b, (short)0, c, false, false);
}

// 32x32 register-blocked GEMM core: 2 A-frags x 2 B-frags -> 4 WMMAs / K-step.
// A rows at A[(m)*lda], B "rows" (columns of the logical B) at Bm[(n)*ldb].
__device__ __forceinline__ void gemm32x32(const bf16_t* __restrict__ A, int lda,
                                          const bf16_t* __restrict__ Bm, int ldb,
                                          int kdim, int lane, v8f acc[4]) {
    for (int kb = 0; kb < kdim; kb += 32) {
        if (kb + 32 < kdim) {   // speculative L2 prefetch of next K-chunk
            __builtin_prefetch(A  + (size_t)(lane & 15) * lda + kb + 32, 0, 1);
            __builtin_prefetch(Bm + (size_t)(lane & 15) * ldb + kb + 32, 0, 1);
        }
        v16bf a0 = frag_selmajor(A, lda, kb, lane);
        v16bf a1 = frag_selmajor(A + (size_t)16 * lda, lda, kb, lane);
        v16bf b0 = frag_selmajor(Bm, ldb, kb, lane);
        v16bf b1 = frag_selmajor(Bm + (size_t)16 * ldb, ldb, kb, lane);
        acc[0] = wmma_bf16(a0, b0, acc[0]);
        acc[1] = wmma_bf16(a0, b1, acc[1]);
        acc[2] = wmma_bf16(a1, b0, acc[2]);
        acc[3] = wmma_bf16(a1, b1, acc[3]);
    }
}

__device__ __forceinline__ void store32x32_bf16(bf16_t* out, size_t ldo,
                                                const v8f acc[4], int lane) {
    int n = lane & 15, mo = ((lane >> 4) & 1) << 3;
#pragma unroll
    for (int v = 0; v < 8; ++v) {
        int m = v + mo;
        out[(size_t)m * ldo + n]           = (bf16_t)acc[0][v];
        out[(size_t)m * ldo + n + 16]      = (bf16_t)acc[1][v];
        out[(size_t)(m + 16) * ldo + n]    = (bf16_t)acc[2][v];
        out[(size_t)(m + 16) * ldo + n + 16] = (bf16_t)acc[3][v];
    }
}

// ---------------------------------------------------------------------------
// f32 -> bf16 conversion (plain and transposing variants)
// ---------------------------------------------------------------------------
__global__ __launch_bounds__(256) void mil_cvt_bf16(const float* __restrict__ s,
                                                    bf16_t* __restrict__ d, int n) {
    int i = blockIdx.x * 256 + threadIdx.x;
    if (i < n) d[i] = (bf16_t)s[i];
}

// relmem [R x RMS] -> rmT_bf [RMS x R]
__global__ __launch_bounds__(256) void mil_cvt_T(const float* __restrict__ s,
                                                 bf16_t* __restrict__ d) {
    int i = blockIdx.x * 256 + threadIdx.x;
    if (i < Rc * RMSc) {
        int r = i / RMSc, j = i % RMSc;
        d[(size_t)j * Rc + r] = (bf16_t)s[i];
    }
}

// ---------------------------------------------------------------------------
// s_w[r, h] = sum_m relmem[r,m] * Wr[h,m] + br[h]    (bf16, natural [R x H])
// 32x32 tile per wave; 1024 tiles -> 128 blocks; K = RMS = 512.
// ---------------------------------------------------------------------------
__global__ __launch_bounds__(256) void mil_gemm_sw(const bf16_t* __restrict__ rm_bf,
                                                   const bf16_t* __restrict__ Wr_bf,
                                                   const float* __restrict__ br,
                                                   bf16_t* __restrict__ s_w) {
    int lane = threadIdx.x & 31, wave = threadIdx.x >> 5;
    int tile = blockIdx.x * 8 + wave;                 // 1024 tiles
    int r0 = (tile >> 5) * 32, h0 = (tile & 31) * 32;
    v8f acc[4] = {{}, {}, {}, {}};
    gemm32x32(rm_bf + (size_t)r0 * RMSc, RMSc,        // A: relmem rows
              Wr_bf + (size_t)h0 * RMSc, RMSc,        // B: Wr^T (sel-major)
              RMSc, lane, acc);
    int n = lane & 15, mo = ((lane >> 4) & 1) << 3;
    float bia0 = br[h0 + n], bia1 = br[h0 + n + 16];
#pragma unroll
    for (int v = 0; v < 8; ++v) {
        int m = v + mo;
        s_w[(size_t)(r0 + m) * Hc + (h0 + n)]        = (bf16_t)(acc[0][v] + bia0);
        s_w[(size_t)(r0 + m) * Hc + (h0 + n + 16)]   = (bf16_t)(acc[1][v] + bia1);
        s_w[(size_t)(r0 + m + 16) * Hc + (h0 + n)]   = (bf16_t)(acc[2][v] + bia0);
        s_w[(size_t)(r0 + m + 16) * Hc + (h0 + n + 16)] = (bf16_t)(acc[3][v] + bia1);
    }
}

// ---------------------------------------------------------------------------
// Gated MIL attention + pooling.  One workgroup per (b,e); 8 waves.
// Clusters (16 x 1024) staged in LDS as bf16.  Per K-step each wave loads ONE
// LDS A-frag and 4 weight B-frags and issues 4 WMMAs (both gates x 2 N-tiles).
// ---------------------------------------------------------------------------
__global__ __launch_bounds__(256) void mil_attn(const float* __restrict__ mention,
                                                const int* __restrict__ entities,
                                                const int* __restrict__ masks,
                                                const bf16_t* __restrict__ Wv_bf,
                                                const bf16_t* __restrict__ Wu_bf,
                                                const float* __restrict__ bv,
                                                const float* __restrict__ bu,
                                                const float* __restrict__ Wa,
                                                const float* __restrict__ ba,
                                                bf16_t* __restrict__ ent_bf) {
    __shared__ bf16_t sClus[Kc][Hc + 8];   // +8 bf16 pad vs LDS bank conflicts
    __shared__ float  sLogit[Kc];
    __shared__ float  sW[Kc];
    __shared__ int    sIdx[Kc];
    __shared__ int    sMask[Kc];

    const int be = blockIdx.x;             // 0..1023
    const int b  = be >> 7;
    const int tid = threadIdx.x, lane = tid & 31, wave = tid >> 5;

    if (tid < Kc) {
        sIdx[tid]   = entities[(size_t)be * Kc + tid];
        sMask[tid]  = masks[(size_t)be * Kc + tid];
        sLogit[tid] = ba[0];
    }
    __syncthreads();

    // Gather K=16 mention rows -> LDS (bf16).
    for (int idx = tid; idx < Kc * Hc; idx += 256) {
        int k = idx >> 10, h = idx & (Hc - 1);
        sClus[k][h] = (bf16_t)mention[((size_t)(b * Mc + sIdx[k])) * Hc + h];
    }
    __syncthreads();

    // Each wave owns a 32-wide slab of the A(=256) dimension.
    const int a0 = wave * 32;
    v8f accV[2] = {{}, {}}, accU[2] = {{}, {}};
    for (int kb = 0; kb < Hc; kb += 32) {
        v16bf af  = frag_selmajor(&sClus[0][0], Hc + 8, kb, lane);            // LDS A
        v16bf fv0 = frag_selmajor(Wv_bf + (size_t)a0 * Hc, Hc, kb, lane);
        v16bf fv1 = frag_selmajor(Wv_bf + (size_t)(a0 + 16) * Hc, Hc, kb, lane);
        v16bf fu0 = frag_selmajor(Wu_bf + (size_t)a0 * Hc, Hc, kb, lane);
        v16bf fu1 = frag_selmajor(Wu_bf + (size_t)(a0 + 16) * Hc, Hc, kb, lane);
        accV[0] = wmma_bf16(af, fv0, accV[0]);
        accV[1] = wmma_bf16(af, fv1, accV[1]);
        accU[0] = wmma_bf16(af, fu0, accU[0]);
        accU[1] = wmma_bf16(af, fu1, accU[1]);
    }
#pragma unroll
    for (int t = 0; t < 2; ++t) {
        int n = a0 + t * 16 + (lane & 15);
        float wa = Wa[n], biasv = bv[n], biasu = bu[n];
#pragma unroll
        for (int v = 0; v < 8; ++v) {
            int m = v + (((lane >> 4) & 1) << 3);          // mention index k
            float gv = tanhf(accV[t][v] + biasv);
            float gu = 1.f / (1.f + expf(-(accU[t][v] + biasu)));
            atomicAdd(&sLogit[m], gv * gu * wa);           // ds_add_f32
        }
    }
    __syncthreads();

    // Masked softmax over K=16 (tiny; single thread).
    if (tid == 0) {
        float l[Kc], mx = NEGV;
#pragma unroll
        for (int k = 0; k < Kc; ++k) { l[k] = sMask[k] ? sLogit[k] : NEGV; mx = fmaxf(mx, l[k]); }
        float sum = 0.f;
#pragma unroll
        for (int k = 0; k < Kc; ++k) { l[k] = expf(l[k] - mx); sum += l[k]; }
        float inv = 1.f / sum;
#pragma unroll
        for (int k = 0; k < Kc; ++k) sW[k] = l[k] * inv;
    }
    __syncthreads();

    // Weighted pooling in f32 from global (L2-resident) rows; emit bf16 row.
    int h0 = tid * 4;
    float4 acc = make_float4(0.f, 0.f, 0.f, 0.f);
#pragma unroll
    for (int k = 0; k < Kc; ++k) {
        float w = sW[k];
        const float4 r = *(const float4*)(mention + ((size_t)(b * Mc + sIdx[k])) * Hc + h0);
        acc.x += w * r.x; acc.y += w * r.y; acc.z += w * r.z; acc.w += w * r.w;
    }
    bf16_t* o = ent_bf + (size_t)be * Hc + h0;
    o[0] = (bf16_t)acc.x; o[1] = (bf16_t)acc.y; o[2] = (bf16_t)acc.z; o[3] = (bf16_t)acc.w;
}

// ---------------------------------------------------------------------------
// s[be, r] = sum_h E[be,h] * s_w[r,h]     (bf16; 1024 tiles -> 128 blocks)
// ---------------------------------------------------------------------------
__global__ __launch_bounds__(256) void mil_gemm_s(const bf16_t* __restrict__ ent_bf,
                                                  const bf16_t* __restrict__ s_w,
                                                  bf16_t* __restrict__ s_bf) {
    int lane = threadIdx.x & 31, wave = threadIdx.x >> 5;
    int tile = blockIdx.x * 8 + wave;
    int be0 = (tile >> 5) * 32, r0 = (tile & 31) * 32;
    v8f acc[4] = {{}, {}, {}, {}};
    gemm32x32(ent_bf + (size_t)be0 * Hc, Hc,          // A: entity rows
              s_w + (size_t)r0 * Hc, Hc,              // B: s_w rows (sel-major)
              Hc, lane, acc);
    store32x32_bf16(s_bf + (size_t)be0 * Rc + r0, Rc, acc, lane);
}

// ---------------------------------------------------------------------------
// s2[be, j] = sum_r s[be,r] * relmem[r,j]  (bf16; 512 tiles -> 64 blocks)
// Uses the transposed copy rmT_bf[RMS x R] so B loads are contiguous.
// ---------------------------------------------------------------------------
__global__ __launch_bounds__(256) void mil_gemm_s2(const bf16_t* __restrict__ s_bf,
                                                   const bf16_t* __restrict__ rmT_bf,
                                                   bf16_t* __restrict__ s2_bf) {
    int lane = threadIdx.x & 31, wave = threadIdx.x >> 5;
    int tile = blockIdx.x * 8 + wave;
    int be0 = (tile >> 4) * 32, j0 = (tile & 15) * 32;
    v8f acc[4] = {{}, {}, {}, {}};
    gemm32x32(s_bf + (size_t)be0 * Rc, Rc,            // A: s rows
              rmT_bf + (size_t)j0 * Rc, Rc,           // B: relmem^T rows
              Rc, lane, acc);
    store32x32_bf16(s2_bf + (size_t)be0 * RMSc + j0, RMSc, acc, lane);
}

// ---------------------------------------------------------------------------
// out[be, h] = sum_j [E | s2][be,j] * Wo[h,j] + bo[h]   (f32 out, K=1536)
// ---------------------------------------------------------------------------
__global__ __launch_bounds__(256) void mil_gemm_out(const bf16_t* __restrict__ ent_bf,
                                                    const bf16_t* __restrict__ s2_bf,
                                                    const bf16_t* __restrict__ Wo_bf,
                                                    const float* __restrict__ bo,
                                                    float* __restrict__ out) {
    int lane = threadIdx.x & 31, wave = threadIdx.x >> 5;
    int tile = blockIdx.x * 8 + wave;                 // 1024 tiles
    int be0 = (tile >> 5) * 32, h0 = (tile & 31) * 32;
    v8f acc[4] = {{}, {}, {}, {}};
    for (int kb = 0; kb < HPc; kb += 32) {            // chunks never straddle j=1024
        const bf16_t* Ap; int lda, kk;
        if (kb < Hc) { Ap = ent_bf + (size_t)be0 * Hc;  lda = Hc;   kk = kb; }
        else         { Ap = s2_bf + (size_t)be0 * RMSc; lda = RMSc; kk = kb - Hc; }
        v16bf a0f = frag_selmajor(Ap, lda, kk, lane);
        v16bf a1f = frag_selmajor(Ap + (size_t)16 * lda, lda, kk, lane);
        v16bf b0f = frag_selmajor(Wo_bf + (size_t)h0 * HPc, HPc, kb, lane);
        v16bf b1f = frag_selmajor(Wo_bf + (size_t)(h0 + 16) * HPc, HPc, kb, lane);
        acc[0] = wmma_bf16(a0f, b0f, acc[0]);
        acc[1] = wmma_bf16(a0f, b1f, acc[1]);
        acc[2] = wmma_bf16(a1f, b0f, acc[2]);
        acc[3] = wmma_bf16(a1f, b1f, acc[3]);
    }
    int n = lane & 15, mo = ((lane >> 4) & 1) << 3;
    float bo0 = bo[h0 + n], bo1 = bo[h0 + n + 16];
#pragma unroll
    for (int v = 0; v < 8; ++v) {
        int m = v + mo;
        out[(size_t)(be0 + m) * Hc + (h0 + n)]           = acc[0][v] + bo0;
        out[(size_t)(be0 + m) * Hc + (h0 + n + 16)]      = acc[1][v] + bo1;
        out[(size_t)(be0 + m + 16) * Hc + (h0 + n)]      = acc[2][v] + bo0;
        out[(size_t)(be0 + m + 16) * Hc + (h0 + n + 16)] = acc[3][v] + bo1;
    }
}

// ---------------------------------------------------------------------------
// Host launcher
// ---------------------------------------------------------------------------
extern "C" void kernel_launch(void* const* d_in, const int* in_sizes, int n_in,
                              void* d_out, int out_size, void* d_ws, size_t ws_size,
                              hipStream_t stream) {
    (void)in_sizes; (void)n_in; (void)out_size; (void)ws_size;

    const float* mention  = (const float*)d_in[0];
    const int*   entities = (const int*)d_in[1];
    const int*   masks    = (const int*)d_in[2];
    const float* relmem   = (const float*)d_in[3];
    const float* Wv = (const float*)d_in[4];
    const float* bv = (const float*)d_in[5];
    const float* Wu = (const float*)d_in[6];
    const float* bu = (const float*)d_in[7];
    const float* Wa = (const float*)d_in[8];
    const float* ba = (const float*)d_in[9];
    const float* Wr = (const float*)d_in[10];
    const float* br = (const float*)d_in[11];
    const float* Wo = (const float*)d_in[12];
    const float* bo = (const float*)d_in[13];
    float* out = (float*)d_out;

    // Workspace layout (bytes, all offsets 512KB-aligned or better)
    char* ws = (char*)d_ws;
    bf16_t* Wv_bf  = (bf16_t*)(ws + 0);              // 256*1024   -> 512KB
    bf16_t* Wu_bf  = (bf16_t*)(ws + (512 << 10));    // 512KB
    bf16_t* Wr_bf  = (bf16_t*)(ws + (1024 << 10));   // 1024*512   -> 1MB
    bf16_t* rm_bf  = (bf16_t*)(ws + (2048 << 10));   // 1024*512   -> 1MB
    bf16_t* rmT_bf = (bf16_t*)(ws + (3072 << 10));   // 512*1024   -> 1MB
    bf16_t* Wo_bf  = (bf16_t*)(ws + (4096 << 10));   // 1024*1536  -> 3MB
    bf16_t* s_w    = (bf16_t*)(ws + (7168 << 10));   // 1024*1024  -> 2MB
    bf16_t* ent_bf = (bf16_t*)(ws + (9216 << 10));   // 1024*1024  -> 2MB
    bf16_t* s_bf   = (bf16_t*)(ws + (11264 << 10));  // 1024*1024  -> 2MB
    bf16_t* s2_bf  = (bf16_t*)(ws + (13312 << 10));  // 1024*512   -> 1MB
    // total 14MB

    // 1) bf16 conversions of weights / relation memory (+ transposed copy)
    mil_cvt_bf16<<<(Ac * Hc + 255) / 256, 256, 0, stream>>>(Wv, Wv_bf, Ac * Hc);
    mil_cvt_bf16<<<(Ac * Hc + 255) / 256, 256, 0, stream>>>(Wu, Wu_bf, Ac * Hc);
    mil_cvt_bf16<<<(Hc * RMSc + 255) / 256, 256, 0, stream>>>(Wr, Wr_bf, Hc * RMSc);
    mil_cvt_bf16<<<(Rc * RMSc + 255) / 256, 256, 0, stream>>>(relmem, rm_bf, Rc * RMSc);
    mil_cvt_T<<<(Rc * RMSc + 255) / 256, 256, 0, stream>>>(relmem, rmT_bf);
    mil_cvt_bf16<<<(Hc * HPc + 255) / 256, 256, 0, stream>>>(Wo, Wo_bf, Hc * HPc);

    // 2) hoisted relation-read projection: s_w = relmem @ Wr^T + br
    mil_gemm_sw<<<128, 256, 0, stream>>>(rm_bf, Wr_bf, br, s_w);

    // 3) gather + gated MIL attention + pooling -> entity_reprs (bf16)
    mil_attn<<<BEc, 256, 0, stream>>>(mention, entities, masks, Wv_bf, Wu_bf,
                                      bv, bu, Wa, ba, ent_bf);

    // 4) relation chain + output projection
    mil_gemm_s<<<128, 256, 0, stream>>>(ent_bf, s_w, s_bf);
    mil_gemm_s2<<<64, 256, 0, stream>>>(s_bf, rmT_bf, s2_bf);
    mil_gemm_out<<<128, 256, 0, stream>>>(ent_bf, s2_bf, Wo_bf, bo, out);
}
